// DSL_19791209300140
// MI455X (gfx1250) — matrix operand order repaired and checked by reading, not compiled, not run on hardware
//
#include <hip/hip_runtime.h>

typedef __attribute__((ext_vector_type(16))) _Float16 v16h;
typedef __attribute__((ext_vector_type(8)))  float    v8f;

#define NROWS 16384
#define DIN   512
#define DH    256        // D/2 = hidden width = K-dim of sim GEMM
#define KNN   8
#define NEG_SLOPE 0.01f
#define CGROUPS 4        // column-split factor for sim kernel
#define SIMP  17         // padded row length for sim tile (bank-conflict-free)

#if __has_builtin(__builtin_amdgcn_global_load_async_to_lds_b128) && \
    __has_builtin(__builtin_amdgcn_s_wait_asynccnt)
#define HAS_ASYNC_LDS 1
// parameter types per hipcc diagnostic: v4i (gcc vector_size) in AS1 / AS3
typedef int Vec4 __attribute__((vector_size(4 * sizeof(int))));
typedef __attribute__((address_space(1))) Vec4 GVec4;
typedef __attribute__((address_space(3))) Vec4 LVec4;
#else
#define HAS_ASYNC_LDS 0
#endif

union Frag {
    v16h   h;
    float4 f4[2];
};

static __device__ __forceinline__ v8f wmma_f16(v16h a, v16h b, v8f c) {
    // 8 args: (neg_a, A, neg_b, B, c_mod, C, reuse_a, reuse_b)
    return __builtin_amdgcn_wmma_f32_16x16x32_f16(false, a, false, b, (short)0, c,
                                                  false, false);
}

// ---------------------------------------------------------------------------
// k0: Wt[n][k] = (f16) W[k][n]   (W is [512, 256] f32 -> Wt [256, 512] f16)
// ---------------------------------------------------------------------------
__global__ void k_wt(const float* __restrict__ W, _Float16* __restrict__ Wt) {
    int idx = blockIdx.x * 256 + threadIdx.x;     // 256*512 = 131072 elems
    int n = idx >> 9;
    int k = idx & 511;
    Wt[n * DIN + k] = (_Float16)W[k * DH + n];
}

// ---------------------------------------------------------------------------
// k1: h = leakyrelu(x@W + b); hn = f16(h / max(||h||, 1e-12))
// block = 256 threads (8 waves), 16 query rows per block, WMMA over K=512.
// ---------------------------------------------------------------------------
__global__ void __launch_bounds__(256) k_fc(const float* __restrict__ x,
                                            const _Float16* __restrict__ Wt,
                                            const float* __restrict__ b,
                                            _Float16* __restrict__ hn) {
    __shared__ __align__(16) _Float16 As[16 * DIN];   // 16 rows of x in f16 (16 KB)
    __shared__ __align__(16) float    Hs[16 * DH];    // post-activation h (16 KB)
    __shared__ float nrm[16];

    const int tid  = threadIdx.x;
    const int lane = tid & 31;
    const int w    = tid >> 5;
    const int rowblk = blockIdx.x * 16;

    // stage + convert A: thread t -> row t/16, 32 contiguous cols
    {
        int r  = tid >> 4;
        int c0 = (tid & 15) * 32;
        const float* xr = x + (size_t)(rowblk + r) * DIN + c0;
        _Float16* as = As + r * DIN + c0;
        #pragma unroll
        for (int i = 0; i < 32; ++i) as[i] = (_Float16)xr[i];
    }
    if (tid < 16) nrm[tid] = 0.f;
    __syncthreads();

    const int arow  = lane & 15;
    const int abase = (lane >= 16) ? 16 : 0;
    const int bkoff = (lane >= 16) ? 16 : 0;
    const int moff  = (lane >= 16) ? 8 : 0;

    // each wave computes two 16x16 output tiles: cols w*32 + {0,16}
    #pragma unroll
    for (int t = 0; t < 2; ++t) {
        const int n = w * 32 + t * 16 + (lane & 15);   // output column 0..255
        v8f c = {};
        for (int kc = 0; kc < DIN; kc += 32) {
            Frag a, bf;
            const float4* ap =
                reinterpret_cast<const float4*>(As + arow * DIN + kc + abase);
            a.f4[0] = ap[0];            // K = kc+base .. +7
            a.f4[1] = ap[1];            // K = kc+base+8 .. +15
            const float4* bp =
                reinterpret_cast<const float4*>(Wt + (size_t)n * DIN + kc + bkoff);
            bf.f4[0] = bp[0];           // K = kc+koff .. +15 (contiguous)
            bf.f4[1] = bp[1];
            c = wmma_f16(a.h, bf.h, c);
        }
        const float bb = b[n];
        #pragma unroll
        for (int r = 0; r < 8; ++r) {
            float h = c[r] + bb;
            h = (h >= 0.f) ? h : NEG_SLOPE * h;
            Hs[(r + moff) * DH + n] = h;
        }
    }
    __syncthreads();

    // row norms via LDS float atomics
    {
        int r  = tid >> 4;
        int c0 = (tid & 15) * 16;
        float s = 0.f;
        #pragma unroll
        for (int i = 0; i < 16; ++i) {
            float h = Hs[r * DH + c0 + i];
            s += h * h;
        }
        atomicAdd(&nrm[r], s);
    }
    __syncthreads();
    {
        int r  = tid >> 4;
        int c0 = (tid & 15) * 16;
        float sc = 1.f / fmaxf(sqrtf(nrm[r]), 1e-12f);
        _Float16* outp = hn + (size_t)(rowblk + r) * DH + c0;
        #pragma unroll
        for (int i = 0; i < 16; ++i)
            outp[i] = (_Float16)(Hs[r * DH + c0 + i] * sc);
    }
}

// ---------------------------------------------------------------------------
// k2: fused sim = hn @ hn.T (f16 WMMA, f32 acc) + running top-8 per query row.
// block = 256 threads (8 waves) = 128 query rows; grid.y splits columns by 4.
// hn (8 MB f16) is L2-resident, so column streaming never hits HBM.
// B tiles staged memory->LDS via GLOBAL_LOAD_ASYNC_TO_LDS_B128 (ASYNCcnt).
// ---------------------------------------------------------------------------
__global__ void __launch_bounds__(256) k_sim_topk(const _Float16* __restrict__ hn,
                                                  float* __restrict__ pval,
                                                  int* __restrict__ pidx) {
    __shared__ __align__(16) _Float16 Bs[16 * DH];    // 16 candidate rows (8 KB)
    __shared__ float simS[8 * 16 * SIMP];             // per-wave padded 16x16 tiles

    const int tid  = threadIdx.x;
    const int lane = tid & 31;
    const int w    = tid >> 5;
    const int qbase = blockIdx.x * 128 + w * 16;
    const int cb    = blockIdx.y * (NROWS / CGROUPS);   // column range base
    const int arow  = lane & 15;
    const int abase = (lane >= 16) ? 16 : 0;
    const int bkoff = (lane >= 16) ? 16 : 0;
    const int moff  = (lane >= 16) ? 8 : 0;
    const int nbase = (lane >= 16) ? 8 : 0;

    // Preload all 8 A fragments (K = 0..255) for this wave's 16 query rows.
    Frag af[8];
    {
        const _Float16* rp = hn + (size_t)(qbase + arow) * DH;
        #pragma unroll
        for (int f = 0; f < 8; ++f) {
            const float4* p =
                reinterpret_cast<const float4*>(rp + f * 32 + abase);
            af[f].f4[0] = p[0];
            af[f].f4[1] = p[1];
        }
    }

    // running top-8 state: lane handles row (lane&15), column half nbase..nbase+7
    float tv[8];
    int   ti[8];
    #pragma unroll
    for (int i = 0; i < 8; ++i) { tv[i] = -1e30f; ti[i] = -1; }
    float mn = -1e30f;
    int   ms = 0;

    const int wtile = w * 16 * SIMP;
    const int srow  = wtile + arow * SIMP;
    const int JT    = (NROWS / CGROUPS) / 16;   // 256 column tiles

    // staging addresses for this thread (row tid/16, 16 halves at (tid%16)*16)
    const int str = tid >> 4;
    const int stc = (tid & 15) * 16;
    _Float16* bdst = Bs + str * DH + stc;

    for (int jt = 0; jt < JT; ++jt) {
        // stage 16 candidate rows into LDS (all waves share them)
        {
            const _Float16* src = hn + (size_t)(cb + jt * 16 + str) * DH + stc;
#if HAS_ASYNC_LDS
            // INST_OFFSET is added to both the global and the LDS address.
            __builtin_amdgcn_global_load_async_to_lds_b128(
                (GVec4*)(void*)src, (LVec4*)(void*)bdst, 0, 0);
            __builtin_amdgcn_global_load_async_to_lds_b128(
                (GVec4*)(void*)src, (LVec4*)(void*)bdst, 16, 0);
            __builtin_amdgcn_s_wait_asynccnt(0);
#else
            const float4* s4 = reinterpret_cast<const float4*>(src);
            float4* d4 = reinterpret_cast<float4*>(bdst);
            d4[0] = s4[0];
            d4[1] = s4[1];
#endif
            if (jt + 1 < JT)   // global_prefetch_b8 for next tile (L2 warm)
                __builtin_prefetch(hn + (size_t)(cb + (jt + 1) * 16 + str) * DH + stc,
                                   0, 1);
        }
        __syncthreads();

        // 16x16 sim tile per wave: 8 WMMAs over K=256
        v8f c = {};
        #pragma unroll
        for (int f = 0; f < 8; ++f) {
            Frag bf;
            const float4* bp = reinterpret_cast<const float4*>(
                Bs + (lane & 15) * DH + f * 32 + bkoff);
            bf.f4[0] = bp[0];
            bf.f4[1] = bp[1];
            c = wmma_f16(af[f].h, bf.h, c);
        }
        #pragma unroll
        for (int r = 0; r < 8; ++r)
            simS[wtile + (r + moff) * SIMP + (lane & 15)] = c[r];
        __syncthreads();

        // top-8 scan: 8 candidates per lane from this tile (conflict-free rows)
        #pragma unroll
        for (int i = 0; i < 8; ++i) {
            float v = simS[srow + nbase + i];
            int   j = cb + jt * 16 + nbase + i;
            if (v > mn) {
                tv[ms] = v;
                ti[ms] = j;
                mn = tv[0]; ms = 0;
                #pragma unroll
                for (int s = 1; s < 8; ++s)
                    if (tv[s] < mn) { mn = tv[s]; ms = s; }
            }
        }
    }

    // merge the two column-halves of each row (lane m <- lane m+16)
    #pragma unroll
    for (int i = 0; i < 8; ++i) {
        float ov = __shfl_down(tv[i], 16, 32);
        int   oi = __shfl_down(ti[i], 16, 32);
        if (lane < 16 && ov > mn) {
            tv[ms] = ov;
            ti[ms] = oi;
            mn = tv[0]; ms = 0;
            #pragma unroll
            for (int s = 1; s < 8; ++s)
                if (tv[s] < mn) { mn = tv[s]; ms = s; }
        }
    }

    if (lane < 16) {
        int q  = qbase + arow;
        int gb = (q * CGROUPS + blockIdx.y) * 8;
        #pragma unroll
        for (int i = 0; i < 8; ++i) {
            pval[gb + i] = tv[i];
            pidx[gb + i] = ti[i];
        }
    }
}

// ---------------------------------------------------------------------------
// k3: merge 4x8 partial candidates -> sorted top-8; emit edge_index + nbr list
// ---------------------------------------------------------------------------
__global__ void k_merge(const float* __restrict__ pval, const int* __restrict__ pidx,
                        int* __restrict__ nbr, float* __restrict__ out_ei) {
    int q = blockIdx.x * 256 + threadIdx.x;
    if (q >= NROWS) return;
    float v[32];
    int   id[32];
    #pragma unroll
    for (int i = 0; i < 32; ++i) {
        v[i]  = pval[q * 32 + i];
        id[i] = pidx[q * 32 + i];
    }
    for (int sel = 0; sel < KNN; ++sel) {
        int bi = 0;
        for (int i = 1; i < 32; ++i) {
            bool better = (v[i] > v[bi]) || (v[i] == v[bi] && id[i] < id[bi]);
            if (better) bi = i;
        }
        int nb = id[bi];
        nbr[q * KNN + sel]    = nb;
        out_ei[q * KNN + sel] = (float)nb;                      // row = neighbor
        out_ei[NROWS * KNN + q * KNN + sel] = (float)q;         // col = query
        v[bi]  = -1e30f;
        id[bi] = 0x7fffffff;
    }
}

// ---------------------------------------------------------------------------
// k4: edge_attr[q] = mean_k x[nbr[q][k]]   (float4, one block per query)
// ---------------------------------------------------------------------------
__global__ void k_edge_attr(const float* __restrict__ x, const int* __restrict__ nbr,
                            float* __restrict__ ea) {
    __shared__ int nb[KNN];
    int q = blockIdx.x;
    if (threadIdx.x < KNN) nb[threadIdx.x] = nbr[q * KNN + threadIdx.x];
    __syncthreads();
    int t = threadIdx.x;                // 0..127, 4 floats each = 512
    float4 acc = make_float4(0.f, 0.f, 0.f, 0.f);
    #pragma unroll
    for (int k = 0; k < KNN; ++k) {
        const float4* xr = reinterpret_cast<const float4*>(x + (size_t)nb[k] * DIN);
        float4 xv = xr[t];
        acc.x += xv.x; acc.y += xv.y; acc.z += xv.z; acc.w += xv.w;
    }
    float4 r = make_float4(acc.x * 0.125f, acc.y * 0.125f,
                           acc.z * 0.125f, acc.w * 0.125f);
    reinterpret_cast<float4*>(ea + (size_t)q * DIN)[t] = r;
}

// ---------------------------------------------------------------------------
// launch
// ---------------------------------------------------------------------------
extern "C" void kernel_launch(void* const* d_in, const int* in_sizes, int n_in,
                              void* d_out, int out_size, void* d_ws, size_t ws_size,
                              hipStream_t stream) {
    const float* x = (const float*)d_in[0];
    const float* W = (const float*)d_in[1];
    const float* b = (const float*)d_in[2];
    float* out = (float*)d_out;

    // workspace layout
    char* ws = (char*)d_ws;
    const size_t WT_OFF   = 0;                                   // 256 KB f16
    const size_t HN_OFF   = WT_OFF + (size_t)DH * DIN * 2;       // 8 MB f16
    const size_t PVAL_OFF = HN_OFF + (size_t)NROWS * DH * 2;     // 2 MB f32
    const size_t PIDX_OFF = PVAL_OFF + (size_t)NROWS * CGROUPS * 8 * 4;
    const size_t NBR_OFF  = PIDX_OFF + (size_t)NROWS * CGROUPS * 8 * 4;

    _Float16* Wt   = (_Float16*)(ws + WT_OFF);
    _Float16* hn   = (_Float16*)(ws + HN_OFF);
    float*    pval = (float*)(ws + PVAL_OFF);
    int*      pidx = (int*)(ws + PIDX_OFF);
    int*      nbr  = (int*)(ws + NBR_OFF);

    // output layout (all float): x | edge_index(2 x N*K) | edge_attr
    float* out_x  = out;
    float* out_ei = out + (size_t)NROWS * DIN;
    float* out_ea = out_ei + 2 * (size_t)NROWS * KNN;

    (void)hipMemcpyAsync(out_x, x, sizeof(float) * (size_t)NROWS * DIN,
                         hipMemcpyDeviceToDevice, stream);

    k_wt<<<(DH * DIN) / 256, 256, 0, stream>>>(W, Wt);
    k_fc<<<NROWS / 16, 256, 0, stream>>>(x, Wt, b, hn);

    dim3 g2(NROWS / 128, CGROUPS);
    k_sim_topk<<<g2, 256, 0, stream>>>(hn, pval, pidx);

    k_merge<<<NROWS / 256, 256, 0, stream>>>(pval, pidx, nbr, out_ei);
    k_edge_attr<<<NROWS, 128, 0, stream>>>(x, nbr, out_ea);
}
